// CGCClass_63479616634913
// MI455X (gfx1250) — matrix operand order repaired and checked by reading, not compiled, not run on hardware
//
#include <hip/hip_runtime.h>
#include <math.h>

#define NNODES  50000
#define NEDGES  640000
#define NGRAPHS 512
#define FEAT    128
#define EDIM    8
#define EHID    64
#define EOUT    32
#define NLAYERS 3
#define DENSE   256
#define OUTC    2
#define EPSV    1e-5f
#define KZ      (FEAT + EOUT)   /* 160 */
#define KTILES  (KZ / 32)       /* 5   */
#define MROWS   32              /* edges per block iteration (2 M-tiles) */

typedef __attribute__((ext_vector_type(16))) __bf16         v16bf;
typedef __attribute__((ext_vector_type(16))) unsigned short v16u;
typedef __attribute__((ext_vector_type(8)))  float          v8f;

__device__ __forceinline__ unsigned short f2bf(float f) {
    unsigned int u = __float_as_uint(f);
    u += 0x7FFFu + ((u >> 16) & 1u);          // round-to-nearest-even
    return (unsigned short)(u >> 16);
}

// fast sigmoid: v_exp_f32 + v_rcp_f32 (no IEEE divide expansion)
__device__ __forceinline__ float sigmoid_fast(float x) {
    return __builtin_amdgcn_rcpf(1.f + __expf(-x));
}

// stable fast softplus: max(v,0) + log(1+exp(-|v|))  (v_exp_f32 + v_log_f32 only)
__device__ __forceinline__ float softplus_fast(float v) {
    return fmaxf(v, 0.f) + __logf(1.f + __expf(-fabsf(v)));
}

__device__ __forceinline__ void atomicMaxF(float* addr, float v) {
    if (v >= 0.f) atomicMax((int*)addr, __float_as_int(v));
    else          atomicMin((unsigned int*)addr, __float_as_uint(v));
}

// ---------------------------------------------------------------- weights->bf16
__global__ void cvt_weights(const float* __restrict__ Wa, const float* __restrict__ Wn,
                            unsigned short* __restrict__ WaBf, unsigned short* __restrict__ WnBf,
                            int n) {
    int i = blockIdx.x * blockDim.x + threadIdx.x;
    if (i < n) { WaBf[i] = f2bf(Wa[i]); WnBf[i] = f2bf(Wn[i]); }
}

// ---------------------------------------------------------------- edge MLP (tiny)
__global__ __launch_bounds__(128)
void edge_mlp(const float* __restrict__ eat,
              const float* __restrict__ We1, const float* __restrict__ be1,
              const float* __restrict__ We2, const float* __restrict__ be2,
              unsigned short* __restrict__ eabf) {
    int e = blockIdx.x * blockDim.x + threadIdx.x;
    if (e >= NEDGES) return;
    float a[EDIM];
    #pragma unroll
    for (int k = 0; k < EDIM; ++k) a[k] = eat[e * EDIM + k];
    float hid[EHID];
    #pragma unroll 4
    for (int h = 0; h < EHID; ++h) {
        float s = be1[h];
        #pragma unroll
        for (int k = 0; k < EDIM; ++k) s += a[k] * We1[k * EHID + h];
        hid[h] = s;
    }
    #pragma unroll 2
    for (int o = 0; o < EOUT; ++o) {
        float s = be2[o];
        #pragma unroll 8
        for (int k = 0; k < EHID; ++k) s += hid[k] * We2[k * EOUT + o];
        eabf[e * EOUT + o] = f2bf(s);
    }
}

// ---------------------------------------------------------------- main WMMA layer kernel
__global__ __launch_bounds__(256, 2)
void layer_edge_wmma(const float* __restrict__ xin,
                     const unsigned short* __restrict__ eabf,
                     const int* __restrict__ dst,
                     const unsigned short* __restrict__ WaBf,   // [160][128] bf16, this layer
                     const unsigned short* __restrict__ WnBf,
                     const float* __restrict__ ba,
                     const float* __restrict__ bnn,
                     float* __restrict__ agg) {
    __shared__ unsigned short zt[MROWS][KZ + 8];
    __shared__ int dstid[MROWS];
    const int tid  = threadIdx.x;
    const int lane = tid & 31;
    const int wave = tid >> 5;
    const int nbase = wave * 16;
    const int ncol  = nbase + (lane & 15);

    // B fragments held in VGPRs for the whole kernel (this code's contiguous k order, same for A:
    // lanes 0-15 K=0..15 of tile, lanes 16-31 K=16..31, N = lane&15).
    v16u bAr[KTILES], bNr[KTILES];
    {
        const int khB = (lane >= 16) ? 16 : 0;
        for (int kt = 0; kt < KTILES; ++kt) {
            v16u ra, rn;
            #pragma unroll
            for (int i = 0; i < 16; ++i) {
                int k = kt * 32 + khB + i;
                ra[i] = WaBf[k * FEAT + ncol];
                rn[i] = WnBf[k * FEAT + ncol];
            }
            bAr[kt] = ra; bNr[kt] = rn;
        }
    }
    const float biasA = ba[ncol];
    const float biasN = bnn[ncol];

    const int numTiles = (NEDGES + MROWS - 1) / MROWS;
    for (int tile = blockIdx.x; tile < numTiles; tile += gridDim.x) {
        // stage z = [x[dst] | ea] as bf16 into LDS (MROWS edges)
        for (int idx = tid; idx < MROWS * KZ; idx += 256) {
            int row = idx / KZ, col = idx - row * KZ;
            int e = tile * MROWS + row;
            unsigned short v = 0;
            if (e < NEDGES) {
                if (col < FEAT) v = f2bf(xin[dst[e] * FEAT + col]);
                else            v = eabf[e * EOUT + (col - FEAT)];
            }
            zt[row][col] = v;
        }
        if (tid < MROWS) {
            int e = tile * MROWS + tid;
            dstid[tid] = (e < NEDGES) ? dst[e] : -1;
        }
        __syncthreads();

        v8f accA0, accN0, accA1, accN1;
        #pragma unroll
        for (int r = 0; r < 8; ++r) {
            accA0[r] = biasA; accN0[r] = biasN;
            accA1[r] = biasA; accN1[r] = biasN;
        }

        const int m0  = lane & 15;          // A rows for M-tile 0
        const int m1  = 16 + m0;            // A rows for M-tile 1
        const int khA = (lane >= 16) ? 8 : 0;
        for (int kt = 0; kt < KTILES; ++kt) {
            // A layout: lanes 0-15 M=0..15, K=0..15 (lanes 16-31 K=16..31), matching B
            v16u ra0, ra1;
            #pragma unroll
            for (int i = 0; i < 16; ++i) {
                int r = i >> 1, w = i & 1;
                int k = kt * 32 + 2 * khA + i;
                ra0[i] = zt[m0][k];
                ra1[i] = zt[m1][k];
            }
            v16bf a0 = __builtin_bit_cast(v16bf, ra0);
            v16bf a1 = __builtin_bit_cast(v16bf, ra1);
            v16bf bA = __builtin_bit_cast(v16bf, bAr[kt]);
            v16bf bN = __builtin_bit_cast(v16bf, bNr[kt]);
            accA0 = __builtin_amdgcn_wmma_f32_16x16x32_bf16(false, a0, false, bA, (short)0, accA0, false, false);
            accN0 = __builtin_amdgcn_wmma_f32_16x16x32_bf16(false, a0, false, bN, (short)0, accN0, false, false);
            accA1 = __builtin_amdgcn_wmma_f32_16x16x32_bf16(false, a1, false, bA, (short)0, accA1, false, false);
            accN1 = __builtin_amdgcn_wmma_f32_16x16x32_bf16(false, a1, false, bN, (short)0, accN1, false, false);
        }

        // m = sigmoid(zWa+ba) * softplus(zWn+bnn); scatter-add into agg (L2-resident)
        #pragma unroll
        for (int r = 0; r < 8; ++r) {
            int mm = r + ((lane >= 16) ? 8 : 0);
            int d0 = dstid[mm];
            if (d0 >= 0) {
                float m = sigmoid_fast(accA0[r]) * softplus_fast(accN0[r]);
                atomicAdd(&agg[d0 * FEAT + ncol], m);
            }
            int d1 = dstid[16 + mm];
            if (d1 >= 0) {
                float m = sigmoid_fast(accA1[r]) * softplus_fast(accN1[r]);
                atomicAdd(&agg[d1 * FEAT + ncol], m);
            }
        }
        __syncthreads();
    }
}

// ---------------------------------------------------------------- batchnorm over nodes
__global__ __launch_bounds__(128)
void bn_stats(const float* __restrict__ agg, float* __restrict__ sum, float* __restrict__ sumsq) {
    int f = threadIdx.x;                       // one feature per lane -> coalesced rows
    float s = 0.f, s2 = 0.f;
    for (int n = blockIdx.x; n < NNODES; n += gridDim.x) {
        float v = agg[n * FEAT + f];
        s += v; s2 += v * v;
    }
    atomicAdd(&sum[f], s);
    atomicAdd(&sumsq[f], s2);
}

__global__ void bn_finalize(const float* __restrict__ sum, const float* __restrict__ sumsq,
                            float* __restrict__ mu, float* __restrict__ rsig, int n, int rows) {
    int f = threadIdx.x;
    if (f < n) {
        float m = sum[f] / (float)rows;
        float v = sumsq[f] / (float)rows - m * m;
        mu[f] = m;
        rsig[f] = rsqrtf(v + EPSV);
    }
}

__global__ void bn_apply_residual(const float* __restrict__ agg,
                                  const float* __restrict__ mu, const float* __restrict__ rsig,
                                  const float* __restrict__ gamma, const float* __restrict__ beta,
                                  float* __restrict__ xcur) {
    int i = blockIdx.x * blockDim.x + threadIdx.x;
    if (i < NNODES * FEAT) {
        int f = i & (FEAT - 1);
        xcur[i] = (agg[i] - mu[f]) * rsig[f] * gamma[f] + beta[f] + xcur[i];
    }
}

// ---------------------------------------------------------------- global max pool
__global__ void pool_init(float* __restrict__ pooled) {
    int i = blockIdx.x * blockDim.x + threadIdx.x;
    if (i < NGRAPHS * FEAT) pooled[i] = __uint_as_float(0xFF800000u);  // -inf
}

__global__ void pool_max(const float* __restrict__ x, const int* __restrict__ batch,
                         float* __restrict__ pooled) {
    int i = blockIdx.x * blockDim.x + threadIdx.x;
    if (i < NNODES * FEAT) {
        int n = i >> 7, f = i & (FEAT - 1);
        atomicMaxF(&pooled[batch[n] * FEAT + f], x[i]);
    }
}

// ---------------------------------------------------------------- dense head (tiny)
__global__ void head1(const float* __restrict__ pooled, const float* __restrict__ W1,
                      const float* __restrict__ b1, float* __restrict__ h) {
    int i = blockIdx.x * blockDim.x + threadIdx.x;
    if (i < NGRAPHS * DENSE) {
        int g = i / DENSE, d = i - g * DENSE;
        float s = b1[d];
        #pragma unroll 8
        for (int k = 0; k < FEAT; ++k) s += pooled[g * FEAT + k] * W1[k * DENSE + d];
        h[i] = fmaxf(s, 0.f);
    }
}

__global__ __launch_bounds__(DENSE)
void head_bn_stats(const float* __restrict__ h, float* __restrict__ mu2, float* __restrict__ rsig2) {
    int f = threadIdx.x;
    float s = 0.f, s2 = 0.f;
    for (int g = 0; g < NGRAPHS; ++g) {
        float v = h[g * DENSE + f];
        s += v; s2 += v * v;
    }
    float m = s / (float)NGRAPHS;
    float v = s2 / (float)NGRAPHS - m * m;
    mu2[f] = m;
    rsig2[f] = rsqrtf(v + EPSV);
}

__global__ void head_out(const float* __restrict__ h,
                         const float* __restrict__ mu2, const float* __restrict__ rsig2,
                         const float* __restrict__ g2, const float* __restrict__ bt2,
                         const float* __restrict__ W2, const float* __restrict__ b2,
                         float* __restrict__ out) {
    int i = blockIdx.x * blockDim.x + threadIdx.x;
    if (i < NGRAPHS * OUTC) {
        int g = i / OUTC, o = i - g * OUTC;
        float s = b2[o];
        #pragma unroll 8
        for (int k = 0; k < DENSE; ++k) {
            float hn = (h[g * DENSE + k] - mu2[k]) * rsig2[k] * g2[k] + bt2[k];
            s += hn * W2[k * OUTC + o];
        }
        out[i] = sigmoid_fast(s);
    }
}

// ---------------------------------------------------------------- launch
extern "C" void kernel_launch(void* const* d_in, const int* in_sizes, int n_in,
                              void* d_out, int out_size, void* d_ws, size_t ws_size,
                              hipStream_t stream) {
    const float* x     = (const float*)d_in[0];
    const int*   eidx  = (const int*)  d_in[1];
    const float* eattr = (const float*)d_in[2];
    const int*   batch = (const int*)  d_in[3];
    const float* We1   = (const float*)d_in[4];
    const float* be1   = (const float*)d_in[5];
    const float* We2   = (const float*)d_in[6];
    const float* be2   = (const float*)d_in[7];
    const float* Wa    = (const float*)d_in[8];
    const float* ba    = (const float*)d_in[9];
    const float* Wn    = (const float*)d_in[10];
    const float* bnn   = (const float*)d_in[11];
    const float* gamma = (const float*)d_in[12];
    const float* beta  = (const float*)d_in[13];
    const float* W1    = (const float*)d_in[14];
    const float* b1    = (const float*)d_in[15];
    const float* g2    = (const float*)d_in[16];
    const float* bt2   = (const float*)d_in[17];
    const float* W2    = (const float*)d_in[18];
    const float* b2    = (const float*)d_in[19];
    const int* dst = eidx + NEDGES;            // edge_index[1]
    float* out = (float*)d_out;

    // workspace bump allocator (~93 MB total)
    char* p = (char*)d_ws;
    auto alloc = [&](size_t bytes) { char* r = p; p += (bytes + 255) & ~(size_t)255; return r; };
    float*          xcur  = (float*)         alloc((size_t)NNODES * FEAT * 4);
    float*          agg   = (float*)         alloc((size_t)NNODES * FEAT * 4);
    unsigned short* eabf  = (unsigned short*)alloc((size_t)NEDGES * EOUT * 2);
    unsigned short* WaBf  = (unsigned short*)alloc((size_t)NLAYERS * KZ * FEAT * 2);
    unsigned short* WnBf  = (unsigned short*)alloc((size_t)NLAYERS * KZ * FEAT * 2);
    float*          bsum  = (float*)alloc(FEAT * 4);
    float*          bsq   = (float*)alloc(FEAT * 4);
    float*          mu    = (float*)alloc(FEAT * 4);
    float*          rsig  = (float*)alloc(FEAT * 4);
    float*          pooled= (float*)alloc((size_t)NGRAPHS * FEAT * 4);
    float*          h     = (float*)alloc((size_t)NGRAPHS * DENSE * 4);
    float*          mu2   = (float*)alloc(DENSE * 4);
    float*          rsig2 = (float*)alloc(DENSE * 4);
    (void)ws_size; (void)n_in; (void)in_sizes; (void)out_size;

    // prep: bf16 weights, working copy of x, edge MLP
    {
        int n = NLAYERS * KZ * FEAT;
        cvt_weights<<<(n + 255) / 256, 256, 0, stream>>>(Wa, Wn, WaBf, WnBf, n);
    }
    hipMemcpyAsync(xcur, x, (size_t)NNODES * FEAT * 4, hipMemcpyDeviceToDevice, stream);
    edge_mlp<<<(NEDGES + 127) / 128, 128, 0, stream>>>(eattr, We1, be1, We2, be2, eabf);

    for (int l = 0; l < NLAYERS; ++l) {
        hipMemsetAsync(agg, 0, (size_t)NNODES * FEAT * 4, stream);
        layer_edge_wmma<<<4096, 256, 0, stream>>>(
            xcur, eabf, dst,
            WaBf + (size_t)l * KZ * FEAT, WnBf + (size_t)l * KZ * FEAT,
            ba + l * FEAT, bnn + l * FEAT, agg);
        hipMemsetAsync(bsum, 0, FEAT * 4, stream);
        hipMemsetAsync(bsq,  0, FEAT * 4, stream);
        bn_stats<<<512, 128, 0, stream>>>(agg, bsum, bsq);
        bn_finalize<<<1, 128, 0, stream>>>(bsum, bsq, mu, rsig, FEAT, NNODES);
        bn_apply_residual<<<(NNODES * FEAT + 255) / 256, 256, 0, stream>>>(
            agg, mu, rsig, gamma + l * FEAT, beta + l * FEAT, xcur);
    }

    pool_init<<<(NGRAPHS * FEAT + 255) / 256, 256, 0, stream>>>(pooled);
    pool_max<<<(NNODES * FEAT + 255) / 256, 256, 0, stream>>>(xcur, batch, pooled);

    head1<<<(NGRAPHS * DENSE + 255) / 256, 256, 0, stream>>>(pooled, W1, b1, h);
    head_bn_stats<<<1, DENSE, 0, stream>>>(h, mu2, rsig2);
    head_out<<<(NGRAPHS * OUTC + 255) / 256, 256, 0, stream>>>(
        h, mu2, rsig2, g2, bt2, W2, b2, out);
}
